// Model_60928406061801
// MI455X (gfx1250) — compile-verified, matrix operand order checked by reading, not run on hardware
//
#include <hip/hip_runtime.h>
#include <hip/hip_bf16.h>

#define LAYERS 3
#define DIM    1024
#define HEADS  16
#define HDIM   64
#define CTX    1024
#define VOCAB  32000
#define BATCH  2
#define BT     (BATCH * CTX)   // 2048

typedef unsigned short u16;
typedef __bf16 v16bf __attribute__((ext_vector_type(16)));
typedef __bf16 v8bf  __attribute__((ext_vector_type(8)));
typedef float  v8f   __attribute__((ext_vector_type(8)));

__device__ __forceinline__ u16 f2bf(float f) {
    union { float f; unsigned u; } c; c.f = f;
    unsigned u = c.u;
    unsigned r = u + 0x7FFFu + ((u >> 16) & 1u);   // round-to-nearest-even
    return (u16)(r >> 16);
}

// WMMA: D = A(16x32 bf16) * B(32x16 bf16) + C(16x16 f32)
__device__ __forceinline__ v8f wmma_bf16(v16bf a, v16bf b, v8f c) {
    return __builtin_amdgcn_wmma_f32_16x16x32_bf16(false, a, false, b, (short)0, c, false, false);
}

// 16-elem bf16 fragment: two contiguous 8-elem K-chunks at off and off+16.
__device__ __forceinline__ v16bf frag_lds(const u16* rowp, int off) {
    union { v8bf h[2]; v16bf v; } r;
    r.h[0] = *(const v8bf*)(rowp + off);
    r.h[1] = *(const v8bf*)(rowp + off + 16);
    return r.v;
}
__device__ __forceinline__ v16bf frag_gbf(const u16* p, int off) {
    union { v8bf h[2]; v16bf v; } r;
    r.h[0] = *(const v8bf*)(p + off);
    r.h[1] = *(const v8bf*)(p + off + 16);
    return r.v;
}

// ---- CDNA5 async global->LDS 16B copy (ASYNCcnt-tracked) ----
__device__ __forceinline__ unsigned lds_off32(const void* p) {
    return (unsigned)(unsigned long long)(size_t)p;   // low 32 bits = LDS byte addr
}
__device__ __forceinline__ void async_b128(unsigned ldsAddr, const void* g) {
    asm volatile("global_load_async_to_lds_b128 %0, %1, off"
                 :: "v"(ldsAddr), "v"((unsigned long long)(size_t)g)
                 : "memory");
}
#if defined(__has_builtin)
#if __has_builtin(__builtin_amdgcn_s_wait_asynccnt)
#define WAIT_ASYNC(n) __builtin_amdgcn_s_wait_asynccnt(n)
#endif
#endif
#ifndef WAIT_ASYNC
#define WAIT_ASYNC(n) asm volatile("s_wait_asynccnt %0" :: "i"(n) : "memory")
#endif

// ---------------------------------------------------------------------------
// Embedding: x[b,t,:] = tok_emb[idx[b,t],:] + pos_emb[t,:]   (f32 out)
// ---------------------------------------------------------------------------
__global__ __launch_bounds__(256) void embed_k(const int* __restrict__ idx,
                                               const float* __restrict__ tok,
                                               const float* __restrict__ pos,
                                               float* __restrict__ x) {
    int row = blockIdx.x, t = threadIdx.x;
    int tid = idx[row];
    int trow = row & (CTX - 1);
    const float* tr = tok + (size_t)tid * DIM;
    const float* pr = pos + (size_t)trow * DIM;
    float* xr = x + (size_t)row * DIM;
#pragma unroll
    for (int i = 0; i < 4; ++i) {
        int d = t + i * 256;
        xr[d] = tr[d] + pr[d];
    }
}

// ---------------------------------------------------------------------------
// LayerNorm (D=1024), f32 in -> bf16 out
// ---------------------------------------------------------------------------
__global__ __launch_bounds__(256) void layernorm_k(const float* __restrict__ x,
                                                   const float* __restrict__ g,
                                                   const float* __restrict__ b,
                                                   u16* __restrict__ out) {
    __shared__ float rs[256], rq[256];
    int row = blockIdx.x, t = threadIdx.x;
    const float* xr = x + (size_t)row * DIM;
    float vals[4]; float s = 0.f, sq = 0.f;
#pragma unroll
    for (int i = 0; i < 4; ++i) {
        float v = xr[t + i * 256];
        vals[i] = v; s += v; sq += v * v;
    }
    rs[t] = s; rq[t] = sq;
    __syncthreads();
    for (int off = 128; off > 0; off >>= 1) {
        if (t < off) { rs[t] += rs[t + off]; rq[t] += rq[t + off]; }
        __syncthreads();
    }
    float mean = rs[0] * (1.0f / DIM);
    float var  = rq[0] * (1.0f / DIM) - mean * mean;
    float inv  = rsqrtf(var + 1e-5f);
    u16* orow = out + (size_t)row * DIM;
#pragma unroll
    for (int i = 0; i < 4; ++i) {
        int d = t + i * 256;
        orow[d] = f2bf((vals[i] - mean) * inv * g[d] + b[d]);
    }
}

// ---------------------------------------------------------------------------
// Weight repack: f32 src(k,n) -> bf16 Wt[n][Kdim], 32x32 LDS tile transpose.
// src(k,n) = W + (n>>nShift)*strideH + k*sK + (n & mask)
//   plain [K][ldb]: nShift=30, strideH=0, sK=ldb
//   qkv  [H][K][HD]: nShift=6, strideH=K*HD, sK=HD
// ---------------------------------------------------------------------------
__global__ __launch_bounds__(256) void repack_k(const float* __restrict__ W,
                                                int nShift, long long strideH, int sK,
                                                u16* __restrict__ Wt, int Kdim) {
    __shared__ float tile[32][33];
    int n0 = blockIdx.x * 32, k0 = blockIdx.y * 32;
    int tx = threadIdx.x & 31, ty = threadIdx.x >> 5;   // 32 x 8
    int n = n0 + tx;
    long long h = (long long)(n >> nShift);
    int nc = n & ((1 << nShift) - 1);
    const float* src = W + h * strideH + nc;
#pragma unroll
    for (int j = 0; j < 4; ++j)
        tile[ty + j * 8][tx] = src[(size_t)(k0 + ty + j * 8) * sK];
    __syncthreads();
#pragma unroll
    for (int j = 0; j < 4; ++j)
        Wt[(size_t)(n0 + ty + j * 8) * Kdim + k0 + tx] = f2bf(tile[tx][ty + j * 8]);
}

// ---------------------------------------------------------------------------
// WMMA GEMM: C[M,N] = A[M,K](bf16) * Bt[N,K](bf16)^T  (+bias)(+res)(relu)
// 128x128 block tile, BK=32, double-buffered async global->LDS staging.
// 8 waves: 4(M) x 2(N); each wave 32x64 = 8 accumulators.
// ---------------------------------------------------------------------------
__global__ __launch_bounds__(256) void gemm_k(
    const u16* __restrict__ A, int lda,
    const u16* __restrict__ Bt, int ldb,
    const float* __restrict__ bias, const float* __restrict__ res,
    void* __restrict__ C, int ldc, int K, int relu, int outBf)
{
    __shared__ alignas(16) u16 As[2][128][40];
    __shared__ alignas(16) u16 Bs[2][128][40];

    int t = threadIdx.x;
    int m0 = blockIdx.y * 128;
    int n0 = blockIdx.x * 128;
    int wave = t >> 5, lane = t & 31;
    int laneHalf = lane >> 4, lane16 = lane & 15;
    int wm = (wave & 3) * 32;
    int wn = (wave >> 2) * 64;

    v8f acc[2][4] = {};

    // staging geometry: 512 16B-chunks per 128x32 tile; thread owns chunks t, t+256
    int r0 = t >> 2;                 // 0..63
    int c0 = (t & 3) * 8;            // 0,8,16,24
    const u16* aBase0 = A  + (size_t)(m0 + r0) * lda + c0;
    const u16* aBase1 = aBase0 + (size_t)64 * lda;
    const u16* bBase0 = Bt + (size_t)(n0 + r0) * ldb + c0;
    const u16* bBase1 = bBase0 + (size_t)64 * ldb;
    unsigned lA0[2] = { lds_off32(&As[0][r0][c0]),      lds_off32(&As[1][r0][c0]) };
    unsigned lA1[2] = { lds_off32(&As[0][r0 + 64][c0]), lds_off32(&As[1][r0 + 64][c0]) };
    unsigned lB0[2] = { lds_off32(&Bs[0][r0][c0]),      lds_off32(&Bs[1][r0][c0]) };
    unsigned lB1[2] = { lds_off32(&Bs[0][r0 + 64][c0]), lds_off32(&Bs[1][r0 + 64][c0]) };

    auto stage = [&](int buf, int k0) {
        async_b128(lA0[buf], aBase0 + k0);
        async_b128(lA1[buf], aBase1 + k0);
        async_b128(lB0[buf], bBase0 + k0);
        async_b128(lB1[buf], bBase1 + k0);
    };

    stage(0, 0);
    int it = 0;
    for (int k0 = 0; k0 < K; k0 += 32, it ^= 1) {
        int hasNext = (k0 + 32) < K;
        if (hasNext) stage(it ^ 1, k0 + 32);   // overlap next transfer with compute
        if (hasNext) WAIT_ASYNC(4); else WAIT_ASYNC(0);
        __syncthreads();                        // all waves' current buffer complete

        int chunk = laneHalf * 8;
        v16bf af[2], bfr[4];
#pragma unroll
        for (int mt = 0; mt < 2; ++mt)
            af[mt] = frag_lds(&As[it][wm + mt * 16 + lane16][0], chunk);
#pragma unroll
        for (int nt = 0; nt < 4; ++nt)
            bfr[nt] = frag_lds(&Bs[it][wn + nt * 16 + lane16][0], chunk);
#pragma unroll
        for (int mt = 0; mt < 2; ++mt)
#pragma unroll
            for (int nt = 0; nt < 4; ++nt)
                acc[mt][nt] = wmma_bf16(af[mt], bfr[nt], acc[mt][nt]);
        __syncthreads();                        // safe to overwrite this buffer
    }

    // epilogue: C-layout row = 8*(lane>>4)+v, col = lane&15
#pragma unroll
    for (int mt = 0; mt < 2; ++mt) {
#pragma unroll
        for (int nt = 0; nt < 4; ++nt) {
            int col = n0 + wn + nt * 16 + lane16;
            float bv = bias ? bias[col] : 0.f;
#pragma unroll
            for (int v = 0; v < 8; ++v) {
                int row = m0 + wm + mt * 16 + laneHalf * 8 + v;
                float val = acc[mt][nt][v] + bv;
                if (res)  val += res[(size_t)row * ldc + col];
                if (relu) val = fmaxf(val, 0.f);
                if (outBf) ((u16*)C)[(size_t)row * ldc + col] = f2bf(val);
                else       ((float*)C)[(size_t)row * ldc + col] = val;
            }
        }
    }
}

// ---------------------------------------------------------------------------
// Flash attention, one (b, h, 64-query tile); 4 waves, wave owns 16 rows.
// q,k,v bf16 [B*T][D] (head slice at col h*HD); out bf16.
// ---------------------------------------------------------------------------
__global__ __launch_bounds__(128) void attn_k(const u16* __restrict__ Q,
                                              const u16* __restrict__ Kp,
                                              const u16* __restrict__ Vp,
                                              u16* __restrict__ O)
{
    __shared__ alignas(16) u16 Plds[4][16][72];
    __shared__ alignas(16) u16 Vt[64][72];

    int qt = blockIdx.x, h = blockIdx.y, b = blockIdx.z;
    int t = threadIdx.x;
    int wave = t >> 5, lane = t & 31;
    int laneHalf = lane >> 4, lane16 = lane & 15;
    int q0 = qt * 64 + wave * 16;

    size_t baseBT = (size_t)b * CTX * DIM;
    const u16* Qb = Q  + baseBT + h * HDIM;
    const u16* Kb = Kp + baseBT + h * HDIM;
    const u16* Vb = Vp + baseBT + h * HDIM;

    v16bf aq[2];
#pragma unroll
    for (int kf = 0; kf < 2; ++kf)
        aq[kf] = frag_gbf(Qb + (size_t)(q0 + lane16) * DIM, kf * 32 + laneHalf * 8);

    const float scale = 0.125f;    // 1/sqrt(64)
    const float NEG = -3.0e38f;
    float m_i[8], l_i[8];
    v8f o[4] = {};
#pragma unroll
    for (int v = 0; v < 8; ++v) { m_i[v] = NEG; l_i[v] = 0.f; }

    for (int kb = 0; kb <= qt; ++kb) {
        v8f s[4] = {};
#pragma unroll
        for (int nt = 0; nt < 4; ++nt) {
            const u16* krow = Kb + (size_t)(kb * 64 + nt * 16 + lane16) * DIM;
#pragma unroll
            for (int kf = 0; kf < 2; ++kf)
                s[nt] = wmma_bf16(aq[kf], frag_gbf(krow, kf * 32 + laneHalf * 8), s[nt]);
        }
        int diag = (kb == qt);
#pragma unroll
        for (int nt = 0; nt < 4; ++nt)
#pragma unroll
            for (int v = 0; v < 8; ++v) {
                float e = s[nt][v] * scale;
                if (diag) {
                    int r = q0 + laneHalf * 8 + v;
                    int sc = kb * 64 + nt * 16 + lane16;
                    if (sc > r) e = NEG;
                }
                s[nt][v] = e;
            }
        float alpha[8];
#pragma unroll
        for (int v = 0; v < 8; ++v) {
            float mx = fmaxf(fmaxf(s[0][v], s[1][v]), fmaxf(s[2][v], s[3][v]));
#pragma unroll
            for (int d = 1; d < 16; d <<= 1) mx = fmaxf(mx, __shfl_xor(mx, d, 32));
            float mnew = fmaxf(m_i[v], mx);
            alpha[v] = __expf(m_i[v] - mnew);
            float su = 0.f;
#pragma unroll
            for (int nt = 0; nt < 4; ++nt) {
                float p = __expf(s[nt][v] - mnew);
                s[nt][v] = p; su += p;
            }
#pragma unroll
            for (int d = 1; d < 16; d <<= 1) su += __shfl_xor(su, d, 32);
            l_i[v] = l_i[v] * alpha[v] + su;
            m_i[v] = mnew;
        }
#pragma unroll
        for (int nt = 0; nt < 4; ++nt)
#pragma unroll
            for (int v = 0; v < 8; ++v) o[nt][v] *= alpha[v];

        // write P (bf16) and stage V tile transposed
#pragma unroll
        for (int nt = 0; nt < 4; ++nt)
#pragma unroll
            for (int v = 0; v < 8; ++v)
                Plds[wave][laneHalf * 8 + v][nt * 16 + lane16] = f2bf(s[nt][v]);
#pragma unroll
        for (int i = 0; i < 4; ++i) {
            int flat8 = i * 128 + t;          // 512 8-elem chunks
            int sl = flat8 >> 3;
            int hd8 = (flat8 & 7) * 8;
            union { v8bf v; u16 u[8]; } tmp;
            tmp.v = *(const v8bf*)(Vb + (size_t)(kb * 64 + sl) * DIM + hd8);
#pragma unroll
            for (int j = 0; j < 8; ++j) Vt[hd8 + j][sl] = tmp.u[j];
        }
        __syncthreads();

        int chunk = laneHalf * 8;
#pragma unroll
        for (int kf = 0; kf < 2; ++kf) {
            v16bf ap = frag_lds(&Plds[wave][lane16][0], kf * 32 + chunk);
#pragma unroll
            for (int nt = 0; nt < 4; ++nt)
                o[nt] = wmma_bf16(ap, frag_lds(&Vt[nt * 16 + lane16][0], kf * 32 + chunk), o[nt]);
        }
        __syncthreads();
    }

#pragma unroll
    for (int nt = 0; nt < 4; ++nt)
#pragma unroll
        for (int v = 0; v < 8; ++v) {
            int r = q0 + laneHalf * 8 + v;
            O[baseBT + (size_t)r * DIM + h * HDIM + nt * 16 + lane16] = f2bf(o[nt][v] / l_i[v]);
        }
}

// ---------------------------------------------------------------------------
extern "C" void kernel_launch(void* const* d_in, const int* in_sizes, int n_in,
                              void* d_out, int out_size, void* d_ws, size_t ws_size,
                              hipStream_t stream) {
    const int*   idx   = (const int*)  d_in[0];
    const float* tok   = (const float*)d_in[1];
    const float* pos   = (const float*)d_in[2];
    const float* wq    = (const float*)d_in[3];
    const float* wk    = (const float*)d_in[4];
    const float* wv    = (const float*)d_in[5];
    const float* wproj = (const float*)d_in[6];
    const float* bproj = (const float*)d_in[7];
    const float* w1    = (const float*)d_in[8];
    const float* b1    = (const float*)d_in[9];
    const float* w2    = (const float*)d_in[10];
    const float* b2    = (const float*)d_in[11];
    const float* ln1g  = (const float*)d_in[12];
    const float* ln1b  = (const float*)d_in[13];
    const float* ln2g  = (const float*)d_in[14];
    const float* ln2b  = (const float*)d_in[15];
    const float* lnfg  = (const float*)d_in[16];
    const float* lnfb  = (const float*)d_in[17];
    const float* wlm   = (const float*)d_in[18];
    const float* blm   = (const float*)d_in[19];

    const size_t NTD = (size_t)BT * DIM;
    const size_t DD  = (size_t)DIM * DIM;
    char* base = (char*)d_ws;
    float* x    = (float*)base;            base += NTD * 4;
    float* sa   = (float*)base;            base += NTD * 4;
    u16*   h    = (u16*)base;              base += NTD * 2;
    u16*   qb   = (u16*)base;              base += NTD * 2;
    u16*   kb   = (u16*)base;              base += NTD * 2;
    u16*   vb   = (u16*)base;              base += NTD * 2;
    u16*   atb  = (u16*)base;              base += NTD * 2;
    u16*   ff1  = (u16*)base;              base += (size_t)BT * 4 * DIM * 2;
    u16*   wscr = (u16*)base;              // max(12*DD, DIM*VOCAB) bf16

    u16* wtq = wscr;
    u16* wtk = wscr + DD;
    u16* wtv = wscr + 2 * DD;
    u16* wtp = wscr + 3 * DD;
    u16* wt1 = wscr + 4 * DD;     // D x 4D
    u16* wt2 = wscr + 8 * DD;     // 4D x D
    u16* wtlm = wscr;             // D x V (after layers)

    const long long qkvSH = (long long)DIM * HDIM;

    embed_k<<<BT, 256, 0, stream>>>(idx, tok, pos, x);

    for (int l = 0; l < LAYERS; ++l) {
        // repack this layer's weights to bf16 [N][K]
        repack_k<<<dim3(DIM / 32, DIM / 32), 256, 0, stream>>>(
            wq + (size_t)l * HEADS * DIM * HDIM, 6, qkvSH, HDIM, wtq, DIM);
        repack_k<<<dim3(DIM / 32, DIM / 32), 256, 0, stream>>>(
            wk + (size_t)l * HEADS * DIM * HDIM, 6, qkvSH, HDIM, wtk, DIM);
        repack_k<<<dim3(DIM / 32, DIM / 32), 256, 0, stream>>>(
            wv + (size_t)l * HEADS * DIM * HDIM, 6, qkvSH, HDIM, wtv, DIM);
        repack_k<<<dim3(DIM / 32, DIM / 32), 256, 0, stream>>>(
            wproj + (size_t)l * DD, 30, 0, DIM, wtp, DIM);
        repack_k<<<dim3(4 * DIM / 32, DIM / 32), 256, 0, stream>>>(
            w1 + (size_t)l * 4 * DD, 30, 0, 4 * DIM, wt1, DIM);
        repack_k<<<dim3(DIM / 32, 4 * DIM / 32), 256, 0, stream>>>(
            w2 + (size_t)l * 4 * DD, 30, 0, DIM, wt2, 4 * DIM);

        layernorm_k<<<BT, 256, 0, stream>>>(x, ln1g + l * DIM, ln1b + l * DIM, h);

        gemm_k<<<dim3(DIM / 128, BT / 128), 256, 0, stream>>>(
            h, DIM, wtq, DIM, nullptr, nullptr, qb, DIM, DIM, 0, 1);
        gemm_k<<<dim3(DIM / 128, BT / 128), 256, 0, stream>>>(
            h, DIM, wtk, DIM, nullptr, nullptr, kb, DIM, DIM, 0, 1);
        gemm_k<<<dim3(DIM / 128, BT / 128), 256, 0, stream>>>(
            h, DIM, wtv, DIM, nullptr, nullptr, vb, DIM, DIM, 0, 1);

        attn_k<<<dim3(CTX / 64, HEADS, BATCH), 128, 0, stream>>>(qb, kb, vb, atb);

        gemm_k<<<dim3(DIM / 128, BT / 128), 256, 0, stream>>>(
            atb, DIM, wtp, DIM, bproj + l * DIM, nullptr, sa, DIM, DIM, 0, 0);

        layernorm_k<<<BT, 256, 0, stream>>>(x, ln2g + l * DIM, ln2b + l * DIM, h);

        gemm_k<<<dim3(4 * DIM / 128, BT / 128), 256, 0, stream>>>(
            h, DIM, wt1, DIM, b1 + (size_t)l * 4 * DIM, nullptr, ff1, 4 * DIM, DIM, 1, 1);

        gemm_k<<<dim3(DIM / 128, BT / 128), 256, 0, stream>>>(
            ff1, 4 * DIM, wt2, 4 * DIM, b2 + l * DIM, sa, x, DIM, 4 * DIM, 0, 0);
    }

    layernorm_k<<<BT, 256, 0, stream>>>(x, lnfg, lnfb, h);

    repack_k<<<dim3(VOCAB / 32, DIM / 32), 256, 0, stream>>>(
        wlm, 30, 0, VOCAB, wtlm, DIM);
    gemm_k<<<dim3(VOCAB / 128, BT / 128), 256, 0, stream>>>(
        h, DIM, wtlm, DIM, blm, nullptr, (float*)d_out, VOCAB, DIM, 0, 0);
}
